// QuantizeEMAResetMultiHead_73985106641321
// MI455X (gfx1250) — matrix operand order repaired and checked by reading, not compiled, not run on hardware
//
#include <hip/hip_runtime.h>
#include <hip/hip_bf16.h>

typedef __attribute__((ext_vector_type(16))) __bf16         v16bf;
typedef __attribute__((ext_vector_type(16))) unsigned short v16u;
typedef __attribute__((ext_vector_type(8)))  unsigned int   v8u;
typedef __attribute__((ext_vector_type(8)))  float          v8f;
typedef __attribute__((ext_vector_type(4)))  unsigned int   u32x4;
typedef __attribute__((ext_vector_type(8)))  int            i32x8;
typedef __attribute__((ext_vector_type(4)))  int            i32x4;

#define NBATCH 32
#define CDIM   512
#define TLEN   2048
#define HEADS  2
#define KCODES 512
#define DHEAD  256
#define TTILE  64
#define RSTR   264   // padded LDS row stride (ushorts): 512B row + 16B pad -> conflict-free b128
#define NTILES (KCODES / 16)

#ifndef __has_builtin
#define __has_builtin(x) 0
#endif
#if __has_builtin(__builtin_amdgcn_tensor_load_to_lds) && __has_builtin(__builtin_amdgcn_s_wait_tensorcnt)
#define USE_TDM 1
#else
#define USE_TDM 0
#endif

__device__ __forceinline__ unsigned short f2bf(float f) {
  unsigned int u = __float_as_uint(f);
  u += 0x7FFFu + ((u >> 16) & 1u);     // round-to-nearest-even
  return (unsigned short)(u >> 16);
}

#if USE_TDM
// ---------------------------------------------------------------------------
// TDM: DMA one 16-code x 256-d bf16 tile (rows of 512B, +16B LDS pad per row)
// D# per CDNA5 ISA ch.8: group0 = {flags, lds_addr, global_addr, type=2},
// group1 = {pad cfg + data_size, tensor dims 256x512, tile 256x16, stride 256}
// ---------------------------------------------------------------------------
__device__ __forceinline__ void tdm_load_tile(const unsigned short* src,
                                              unsigned int lds_off) {
  unsigned long long ga = (unsigned long long)(uintptr_t)src;
  u32x4 g0;
  g0[0] = 1u;                                                   // count=1 (valid), user mode
  g0[1] = lds_off;                                              // lds_addr[31:0]
  g0[2] = (unsigned int)ga;                                     // global_addr[31:0]
  g0[3] = (unsigned int)((ga >> 32) & 0x01FFFFFFull)            // global_addr[56:32]
        | (2u << 30);                                           // type = 2 ("image")
  i32x8 g1;
  g1[0] = (int)((1u << 16)      // data_size = 1 -> 2-byte elements
              | (1u << 20)      // pad_enable
              | (6u << 22)      // pad_interval: 128 DWORDs (512B) between pads
              | (3u << 25));    // pad_amount: 4 DWORDs (16B)
  g1[1] = (int)(256u << 16);    // tensor_dim0[15:0] in bits[63:48]  (= 256 elems/row)
  g1[2] = (int)(512u << 16);    // tensor_dim1[15:0] in bits[95:80]  (= 512 rows)
  g1[3] = (int)(256u << 16);    // tile_dim0 in bits[127:112]        (= 256)
  g1[4] = 16;                   // tile_dim1 in bits[143:128]        (= 16 rows)
  g1[5] = 256;                  // tensor_dim0_stride[31:0]          (= 256 elems)
  g1[6] = 0;
  g1[7] = 0;
  i32x4 z4 = {0, 0, 0, 0};
#if __clang_major__ >= 23
  i32x8 z8 = {0, 0, 0, 0, 0, 0, 0, 0};
  __builtin_amdgcn_tensor_load_to_lds(g0, g1, z4, z4, z8, 0);
#else
  __builtin_amdgcn_tensor_load_to_lds(g0, g1, z4, z4, 0);
#endif
}
#else
__device__ __forceinline__ void copy_tile(const unsigned short* __restrict__ src,
                                          unsigned short* __restrict__ dst, int tid) {
  int row = tid >> 3, c8 = tid & 7;
  const uint4* srow = (const uint4*)(src + (size_t)row * DHEAD);
  uint4* drow = (uint4*)(dst + row * RSTR);
#pragma unroll
  for (int q = 0; q < 4; ++q)
    drow[c8 + q * 8] = srow[c8 + q * 8];
}
#endif

// ---------------------------------------------------------------------------
// Kernel 1: codebook f32 -> bf16, per-code squared norms, zero hist/loss
// ---------------------------------------------------------------------------
__global__ void __launch_bounds__(256)
vq_prep(const float* __restrict__ cb, unsigned short* __restrict__ cb16,
        float* __restrict__ cnorm, unsigned int* __restrict__ hist,
        float* __restrict__ loss_acc)
{
  __shared__ float sd[256];
  const int b   = blockIdx.x;          // flat code id: h*512 + k
  const int tid = threadIdx.x;         // d
  float v = cb[(size_t)b * DHEAD + tid];
  cb16[(size_t)b * DHEAD + tid] = f2bf(v);
  sd[tid] = v * v;
  __syncthreads();
  for (int s = 128; s > 0; s >>= 1) {
    if (tid < s) sd[tid] += sd[tid + s];
    __syncthreads();
  }
  if (tid == 0) cnorm[b] = sd[0];
  hist[(size_t)b * 256 + tid] = 0u;    // 1024*256 == 262144 bins
  if (b == 0 && tid == 0) loss_acc[0] = 0.f;
}

// ---------------------------------------------------------------------------
// Kernel 2: fused distance-GEMM (bf16 WMMA) + argmin + gather + STE out +
//           commit-loss + histogram.  128 threads = 4 waves, each wave owns
//           a 16-row (t) M-tile; grid = (TLEN/TTILE, NBATCH).
//           Codebook tiles are double-buffered in LDS via the Tensor Data
//           Mover (TENSORcnt) so the DMA for tile ct+1 overlaps WMMA on ct.
// ---------------------------------------------------------------------------
__global__ void __launch_bounds__(128)
vq_main(const float* __restrict__ x, const float* __restrict__ cb,
        const unsigned short* __restrict__ cb16, const float* __restrict__ cnorm,
        unsigned int* __restrict__ hist, float* __restrict__ loss_acc,
        float* __restrict__ out)
{
  __shared__ __align__(32) unsigned short Btile[2 * 16 * RSTR]; // double buffer
  __shared__ int   idx_sh[HEADS][TTILE];
  __shared__ float red[128];

  const int tid  = threadIdx.x;
  const int lane = tid & 31;
  const int wave = tid >> 5;
  const int m    = lane & 15;          // M row within wave tile / N col of C tile
  const int kh   = lane >> 4;          // lane half (K split per ISA layout)

  const int n  = blockIdx.y;
  const int t0 = blockIdx.x * TTILE;
  const int t  = t0 + wave * 16 + m;   // this lane's time index (A-matrix row)

#if USE_TDM
#define TILE_LOAD(ct_, buf_)                                                     \
  do { if (wave == 0)                                                            \
         tdm_load_tile(cb16 + (size_t)(h * KCODES + (ct_) * 16) * DHEAD,         \
                       (unsigned int)(uintptr_t)(Btile + (buf_) * (16 * RSTR))); \
  } while (0)
#define TILE_WAIT() do { if (wave == 0) __builtin_amdgcn_s_wait_tensorcnt(0); } while (0)
#else
#define TILE_LOAD(ct_, buf_)                                                     \
  copy_tile(cb16 + (size_t)(h * KCODES + (ct_) * 16) * DHEAD,                    \
            Btile + (buf_) * (16 * RSTR), tid)
#define TILE_WAIT() ((void)0)
#endif

  for (int h = 0; h < HEADS; ++h) {
    // ---- preload A fragments: x[n, h*256 + d, t] as bf16, contiguous k = kh*16 + j (same as B) ----
    const float* xh = x + (size_t)n * CDIM * TLEN + (size_t)h * DHEAD * TLEN + t;
    v16bf afrag[8];
#pragma unroll
    for (int ks = 0; ks < 8; ++ks) {
      v16u au;
#pragma unroll
      for (int j = 0; j < 16; ++j) {
        int d = ks * 32 + kh * 16 + j;
        au[j] = f2bf(xh[(size_t)d * TLEN]);
      }
      afrag[ks] = __builtin_bit_cast(v16bf, au);
    }

    float mind[8];
    int   midx[8];
#pragma unroll
    for (int r = 0; r < 8; ++r) { mind[r] = 3.4e38f; midx[r] = 0; }

    TILE_LOAD(0, 0);                              // prologue DMA for tile 0
    for (int ct = 0; ct < NTILES; ++ct) {
      TILE_WAIT();                                // tile ct has landed in LDS
      __syncthreads();
      if (ct + 1 < NTILES)
        TILE_LOAD(ct + 1, (ct + 1) & 1);          // overlap next DMA with WMMA

      const unsigned short* Bt = Btile + (ct & 1) * (16 * RSTR);
      v8f acc = {0.f, 0.f, 0.f, 0.f, 0.f, 0.f, 0.f, 0.f};
#pragma unroll
      for (int ks = 0; ks < 8; ++ks) {
        const unsigned short* bp = Bt + m * RSTR + ks * 32 + kh * 16;
        uint4 b0 = *(const uint4*)bp;
        uint4 b1 = *(const uint4*)(bp + 8);
        v8u bu = {b0.x, b0.y, b0.z, b0.w, b1.x, b1.y, b1.z, b1.w};
        v16bf bfrag = __builtin_bit_cast(v16bf, bu);
        acc = __builtin_amdgcn_wmma_f32_16x16x32_bf16(
            false, afrag[ks], false, bfrag, (short)0, acc, false, false);
      }

      // dist = ||c||^2 - 2*x.c  (||x||^2 constant over k -> dropped)
      float cn   = cnorm[h * KCODES + ct * 16 + m];
      int   kidx = ct * 16 + m;
#pragma unroll
      for (int r = 0; r < 8; ++r) {
        float dist = __builtin_fmaf(-2.f, acc[r], cn);
        if (dist < mind[r]) { mind[r] = dist; midx[r] = kidx; }
      }
      __syncthreads();   // all waves done reading buf[ct&1] before it is reused
    }

    // cross-lane argmin over the 16 candidate columns (tie -> lowest index)
#pragma unroll
    for (int msk = 1; msk < 16; msk <<= 1) {
#pragma unroll
      for (int r = 0; r < 8; ++r) {
        float ov = __shfl_xor(mind[r], msk, 32);
        int   oi = __shfl_xor(midx[r], msk, 32);
        if (ov < mind[r] || (ov == mind[r] && oi < midx[r])) { mind[r] = ov; midx[r] = oi; }
      }
    }
    if (m == 0) {
#pragma unroll
      for (int r = 0; r < 8; ++r)
        idx_sh[h][wave * 16 + kh * 8 + r] = midx[r];
    }
    __syncthreads();
  }

  // ---- histogram of combined overall indices (one entry per (n,t)) ----
  if (tid < TTILE) {
    int overall = idx_sh[0][tid] + KCODES * idx_sh[1][tid];
    atomicAdd(&hist[overall], 1u);
  }

  // ---- STE output (coalesced along t) + commit loss ----
  const int tl = tid & 63;
  const int tt = t0 + tl;
  float lsum = 0.f;
  for (int c = (tid >> 6); c < CDIM; c += 2) {
    int h = c >> 8, d = c & 255;
    int k = idx_sh[h][tl];
    float q = cb[(size_t)(h * KCODES + k) * DHEAD + d];   // f32 codebook, L2-hot
    size_t off = (size_t)n * CDIM * TLEN + (size_t)c * TLEN + tt;
    float xv = x[off];
    out[off] = xv + (q - xv);                             // matches jnp STE arithmetic
    float df = xv - q;
    lsum = __builtin_fmaf(df, df, lsum);
  }
  red[tid] = lsum;
  __syncthreads();
  for (int s = 64; s > 0; s >>= 1) {
    if (tid < s) red[tid] += red[tid + s];
    __syncthreads();
  }
  if (tid == 0) atomicAdd(loss_acc, red[0]);
#undef TILE_LOAD
#undef TILE_WAIT
}

// ---------------------------------------------------------------------------
// Kernel 3: perplexity from histogram + loss normalization
// ---------------------------------------------------------------------------
__global__ void __launch_bounds__(256)
vq_finalize(const unsigned int* __restrict__ hist,
            const float* __restrict__ loss_acc,
            float* __restrict__ out_scalars)
{
  __shared__ float sd[256];
  const int tid = threadIdx.x;
  const float invN = 1.f / (float)(NBATCH * TLEN);        // 1/65536
  float s = 0.f;
  for (int i = tid; i < KCODES * KCODES; i += 256) {
    float p = (float)hist[i] * invN;
    s += p * __logf(p + 1e-7f);
  }
  sd[tid] = s;
  __syncthreads();
  for (int k = 128; k > 0; k >>= 1) {
    if (tid < k) sd[tid] += sd[tid + k];
    __syncthreads();
  }
  if (tid == 0) {
    out_scalars[0] = loss_acc[0] / (float)((size_t)NBATCH * CDIM * TLEN);
    out_scalars[1] = __expf(-sd[0]);
  }
}

// ---------------------------------------------------------------------------
extern "C" void kernel_launch(void* const* d_in, const int* in_sizes, int n_in,
                              void* d_out, int out_size, void* d_ws, size_t ws_size,
                              hipStream_t stream) {
  (void)in_sizes; (void)n_in; (void)out_size; (void)ws_size;
  const float* x  = (const float*)d_in[0];   // [32, 512, 2048]
  const float* cb = (const float*)d_in[1];   // [2, 512, 256]
  float* out = (float*)d_out;                // 32*512*2048 floats + loss + perplexity

  // workspace layout (~1.52 MB total):
  char* ws = (char*)d_ws;
  float*          loss_acc = (float*)ws;                                   // @0
  float*          cnorm    = (float*)(ws + 1024);                          // 1024 f32
  unsigned int*   hist     = (unsigned int*)(ws + 8192);                   // 262144 u32 (1 MB)
  unsigned short* cb16     = (unsigned short*)(ws + 8192 + (size_t)KCODES * KCODES * 4);

  hipLaunchKernelGGL(vq_prep, dim3(HEADS * KCODES), dim3(256), 0, stream,
                     cb, cb16, cnorm, hist, loss_acc);
  hipLaunchKernelGGL(vq_main, dim3(TLEN / TTILE, NBATCH), dim3(128), 0, stream,
                     x, cb, cb16, cnorm, hist, loss_acc, out);
  hipLaunchKernelGGL(vq_finalize, dim3(1), dim3(256), 0, stream,
                     hist, loss_acc, out + (size_t)NBATCH * CDIM * TLEN);
}
